// GraphTransformerEncoder_6416681140224
// MI455X (gfx1250) — matrix operand order repaired and checked by reading, not compiled, not run on hardware
//
#include <hip/hip_runtime.h>

typedef __attribute__((ext_vector_type(16))) _Float16 v16h;
typedef __attribute__((ext_vector_type(8)))  _Float16 v8h;
typedef __attribute__((ext_vector_type(8)))  float    v8f;

#define HID 128
#define HEADS 4
#define NEG_SLOPE 0.2f
#define LN_EPS 1e-5f

// ---------------- conversion / utility kernels ----------------
__global__ void k_f32_to_f16(const float* __restrict__ s, _Float16* __restrict__ d, int n) {
  int i = blockIdx.x * blockDim.x + threadIdx.x;
  if (i < n) d[i] = (_Float16)s[i];
}

// W[k*C + c] (K x C, row-major)  ->  WT[c*K + k] (C x K, f16)
__global__ void k_convert_transpose(const float* __restrict__ W, _Float16* __restrict__ WT,
                                    int K, int C) {
  int i = blockIdx.x * blockDim.x + threadIdx.x;
  if (i < K * C) {
    int k = i / C, c = i % C;
    WT[c * K + k] = (_Float16)W[i];
  }
}

__global__ void k_copy_f32x4(const float4* __restrict__ a, float4* __restrict__ b, int n4) {
  int i = blockIdx.x * blockDim.x + threadIdx.x;
  if (i < n4) b[i] = a[i];
}

__global__ void k_init_mz4(float4* __restrict__ m, float4* __restrict__ z, int n4) {
  int i = blockIdx.x * blockDim.x + threadIdx.x;
  if (i < n4) {
    m[i] = make_float4(-3.0e38f, -3.0e38f, -3.0e38f, -3.0e38f);
    z[i] = make_float4(0.f, 0.f, 0.f, 0.f);
  }
}

// ---------------- WMMA GEMM: out[M,128] = A16[M,K] * (BT16[128,K])^T + bias ----------------
// 1 wave per 16x16 tile; 8 waves/block cover all 128 output columns.
// WRITE16: also emit an f16 copy of the output (fused conversion for the next GEMM).
template <int K, bool WRITE16>
__global__ void k_gemm_wmma(const _Float16* __restrict__ A,    // [M,K] row-major f16
                            const _Float16* __restrict__ BT,   // [128,K] = W^T, f16
                            const float*    __restrict__ bias, // [128]
                            float*          __restrict__ out,  // [M,128]
                            _Float16*       __restrict__ out16,// [M,128] (if WRITE16)
                            int M) {
  const int lane   = threadIdx.x & 31;
  const int wave   = threadIdx.x >> 5;   // 0..7 -> column tile
  const int l16    = lane & 15;
  const int hiHalf = lane >> 4;          // 0 for lanes 0-15, 1 for lanes 16-31
  const int rowBase = blockIdx.x * 16;
  int row = rowBase + l16; if (row >= M) row = M - 1;   // clamp (dup loads, stores guarded)
  const int col = wave * 16 + l16;

  const _Float16* arow = A  + (size_t)row * K;
  const _Float16* bcol = BT + (size_t)col * K;
  const int aoff = hiHalf * 8;    // A: lanes16-31 hold K base +8 within each 32-slab
  const int boff = hiHalf * 8;    // B: lanes16-31 hold K base +8 within each 32-slab

  v8f acc = {};
#pragma unroll
  for (int k0 = 0; k0 < K; k0 += 32) {
    v8h alo = *(const v8h*)(arow + k0 + aoff);        // K = base .. base+7
    v8h ahi = *(const v8h*)(arow + k0 + aoff + 16);   // K = base+16 .. base+23
    v16h a;
#pragma unroll
    for (int i = 0; i < 8; ++i) { a[i] = alo[i]; a[i + 8] = ahi[i]; }
    v8h blo = *(const v8h*)(bcol + k0 + boff);
    v8h bhi = *(const v8h*)(bcol + k0 + boff + 16);
    v16h b;
#pragma unroll
    for (int i = 0; i < 8; ++i) { b[i] = blo[i]; b[i + 8] = bhi[i]; }
    acc = __builtin_amdgcn_wmma_f32_16x16x32_f16(false, a, false, b,
                                                 (short)0, acc, false, false);
  }
  const float bv = bias[col];
  // C/D layout: VGPR r -> row (rowBase + hiHalf*8 + r), column = col
  const size_t obase = (size_t)(rowBase + hiHalf * 8) * HID + col;
  float*    op  = out + obase;
  _Float16* op16 = WRITE16 ? (out16 + obase) : nullptr;
  if (rowBase + 16 <= M) {            // fast path: branch-free, immediate-offset stores
#pragma unroll
    for (int r = 0; r < 8; ++r) {
      float v = acc[r] + bv;
      op[(size_t)r * HID] = v;
      if (WRITE16) op16[(size_t)r * HID] = (_Float16)v;
    }
  } else {                            // ragged tail (never taken when M % 16 == 0)
#pragma unroll
    for (int r = 0; r < 8; ++r) {
      int orow = rowBase + hiHalf * 8 + r;
      if (orow < M) {
        float v = acc[r] + bv;
        out[(size_t)orow * HID + col] = v;
        if (WRITE16) out16[(size_t)orow * HID + col] = (_Float16)v;
      }
    }
  }
}

// ---------------- edge kernels (wave per edge, float4 per lane) ----------------
__device__ __forceinline__ void atomicMaxF(float* a, float v) {
  unsigned int* ua = (unsigned int*)a;
  unsigned int old = __float_as_uint(*a);
  while (__uint_as_float(old) < v) {
    unsigned int assumed = old;
    old = atomicCAS(ua, assumed, __float_as_uint(v));
    if (old == assumed) break;
  }
}

__global__ void k_edge_logits(const float* __restrict__ fs, const float* __restrict__ fd,
                              const float* __restrict__ attn,   // [128] = [H][DH]
                              const int* __restrict__ src, const int* __restrict__ dst,
                              float* __restrict__ logits,       // [EP,H]
                              float* __restrict__ m,            // [N,H]
                              int E, int EP) {
  const int lane = threadIdx.x & 31;
  const int e = blockIdx.x * 8 + (threadIdx.x >> 5);
  if (e >= EP) return;
  const int s = (e < E) ? src[e] : (e - E);
  const int d = (e < E) ? dst[e] : (e - E);
  const int c = lane * 4;                       // dims [c..c+3]; head = c/32 = lane/8
  float4 a4 = *(const float4*)(attn + c);
  float4 x  = *(const float4*)(fs + (size_t)s * HID + c);
  float4 y  = *(const float4*)(fd + (size_t)d * HID + c);
  float v, p = 0.0f;
  v = x.x + y.x; v = v > 0.f ? v : NEG_SLOPE * v; p += a4.x * v;
  v = x.y + y.y; v = v > 0.f ? v : NEG_SLOPE * v; p += a4.y * v;
  v = x.z + y.z; v = v > 0.f ? v : NEG_SLOPE * v; p += a4.z * v;
  v = x.w + y.w; v = v > 0.f ? v : NEG_SLOPE * v; p += a4.w * v;
  p += __shfl_xor(p, 1, 8);
  p += __shfl_xor(p, 2, 8);
  p += __shfl_xor(p, 4, 8);
  if ((lane & 7) == 0) {
    const int hd = lane >> 3;
    logits[(size_t)e * HEADS + hd] = p;
    atomicMaxF(m + d * HEADS + hd, p);
  }
}

// one thread per edge: softmax numerator for all 4 heads (float4)
__global__ void k_edge_exp(float4* __restrict__ logits, const float4* __restrict__ m,
                           float* __restrict__ z, const int* __restrict__ dst,
                           int E, int EP) {
  int e = blockIdx.x * blockDim.x + threadIdx.x;
  if (e >= EP) return;
  const int d = (e < E) ? dst[e] : (e - E);
  float4 lg = logits[e];
  float4 mv = m[d];
  float4 p = make_float4(__expf(lg.x - mv.x), __expf(lg.y - mv.y),
                         __expf(lg.z - mv.z), __expf(lg.w - mv.w));
  logits[e] = p;
  float* zd = z + (size_t)d * HEADS;
  atomicAdd(zd + 0, p.x);
  atomicAdd(zd + 1, p.y);
  atomicAdd(zd + 2, p.z);
  atomicAdd(zd + 3, p.w);
}

__global__ void k_edge_scatter(const float* __restrict__ fs, const float* __restrict__ p,
                               const float* __restrict__ z, const int* __restrict__ src,
                               const int* __restrict__ dst, float* __restrict__ agg,
                               int E, int EP) {
  const int lane = threadIdx.x & 31;
  const int e = blockIdx.x * 8 + (threadIdx.x >> 5);
  if (e >= EP) return;
  const int s = (e < E) ? src[e] : (e - E);
  const int d = (e < E) ? dst[e] : (e - E);
  const int hd = lane >> 3;
  const float alpha = p[(size_t)e * HEADS + hd] / z[(size_t)d * HEADS + hd];
  const int c = lane * 4;
  float4 x = *(const float4*)(fs + (size_t)s * HID + c);
  float* o = agg + (size_t)d * HID + c;
  atomicAdd(o + 0, x.x * alpha);
  atomicAdd(o + 1, x.y * alpha);
  atomicAdd(o + 2, x.z * alpha);
  atomicAdd(o + 3, x.w * alpha);
}

// ---------------- LayerNorm(h + h_new) + ReLU, wave per row ----------------
// Also emits the f16 copy consumed by the next layer's WMMA GEMMs.
__global__ void k_ln_relu(float* __restrict__ h, const float* __restrict__ agg,
                          const float* __restrict__ g, const float* __restrict__ b,
                          _Float16* __restrict__ h16, int N) {
  const int lane = threadIdx.x & 31;
  const int row = blockIdx.x * 8 + (threadIdx.x >> 5);
  if (row >= N) return;
  const int c = lane * 4;
  float4 hv = *(const float4*)(h   + (size_t)row * HID + c);
  float4 av = *(const float4*)(agg + (size_t)row * HID + c);
  float x0 = hv.x + av.x, x1 = hv.y + av.y, x2 = hv.z + av.z, x3 = hv.w + av.w;
  float s = x0 + x1 + x2 + x3;
  float q = x0 * x0 + x1 * x1 + x2 * x2 + x3 * x3;
#pragma unroll
  for (int o = 1; o < 32; o <<= 1) { s += __shfl_xor(s, o, 32); q += __shfl_xor(q, o, 32); }
  const float mu = s * (1.0f / HID);
  const float var = q * (1.0f / HID) - mu * mu;
  const float rs = rsqrtf(var + LN_EPS);
  float4 gv = *(const float4*)(g + c);
  float4 bv = *(const float4*)(b + c);
  float y0 = (x0 - mu) * rs * gv.x + bv.x; y0 = y0 > 0.f ? y0 : 0.f;
  float y1 = (x1 - mu) * rs * gv.y + bv.y; y1 = y1 > 0.f ? y1 : 0.f;
  float y2 = (x2 - mu) * rs * gv.z + bv.z; y2 = y2 > 0.f ? y2 : 0.f;
  float y3 = (x3 - mu) * rs * gv.w + bv.w; y3 = y3 > 0.f ? y3 : 0.f;
  *(float4*)(h + (size_t)row * HID + c) = make_float4(y0, y1, y2, y3);
  _Float16* o16 = h16 + (size_t)row * HID + c;
  o16[0] = (_Float16)y0; o16[1] = (_Float16)y1;
  o16[2] = (_Float16)y2; o16[3] = (_Float16)y3;
}

// ---------------- graph mean (128 columns) ----------------
__global__ void k_colmean(const float* __restrict__ h, float* __restrict__ out, int N) {
  __shared__ float red[256];
  const int c = blockIdx.x;
  float s = 0.0f;
  for (int r = threadIdx.x; r < N; r += blockDim.x) s += h[(size_t)r * HID + c];
  red[threadIdx.x] = s;
  __syncthreads();
  for (int o = 128; o > 0; o >>= 1) {
    if ((int)threadIdx.x < o) red[threadIdx.x] += red[threadIdx.x + o];
    __syncthreads();
  }
  if (threadIdx.x == 0) out[c] = red[0] / (float)N;
}

// ---------------- host-side orchestration ----------------
extern "C" void kernel_launch(void* const* d_in, const int* in_sizes, int n_in,
                              void* d_out, int out_size, void* d_ws, size_t ws_size,
                              hipStream_t stream) {
  const float* node_feats = (const float*)d_in[0];
  const int*   src        = (const int*)d_in[1];
  const int*   dst        = (const int*)d_in[2];
  const float* W_in       = (const float*)d_in[3];
  const float* b_in       = (const float*)d_in[4];
  const float* W_src      = (const float*)d_in[5];
  const float* b_src      = (const float*)d_in[6];
  const float* W_dst      = (const float*)d_in[7];
  const float* b_dst      = (const float*)d_in[8];
  const float* attn       = (const float*)d_in[9];
  const float* ln_g       = (const float*)d_in[10];
  const float* ln_b       = (const float*)d_in[11];
  float* out = (float*)d_out;

  const int IN = 64;
  const int N  = in_sizes[0] / IN;
  const int E  = in_sizes[1];
  const int EP = E + N;

  // ---- carve workspace ----
  char* w = (char*)d_ws;
  auto carve = [&](size_t bytes) -> void* {
    void* p = (void*)w;
    w += (bytes + 255) & ~(size_t)255;
    return p;
  };
  float*    h      = (float*)carve((size_t)N * HID * 4);
  float*    fs     = (float*)carve((size_t)N * HID * 4);
  float*    fd     = (float*)carve((size_t)N * HID * 4);
  float*    agg    = (float*)carve((size_t)N * HID * 4);
  float*    mbuf   = (float*)carve((size_t)N * HEADS * 4);
  float*    zbuf   = (float*)carve((size_t)N * HEADS * 4);
  float*    logits = (float*)carve((size_t)EP * HEADS * 4);
  _Float16* h16    = (_Float16*)carve((size_t)N * HID * 2);
  _Float16* x16    = (_Float16*)carve((size_t)N * IN * 2);
  _Float16* WTin   = (_Float16*)carve((size_t)HID * IN * 2);
  _Float16* WsT    = (_Float16*)carve((size_t)3 * HID * HID * 2);
  _Float16* WdT    = (_Float16*)carve((size_t)3 * HID * HID * 2);

  const int T = 256;
  auto cdiv = [](int a, int b) { return (a + b - 1) / b; };

  // ---- weights -> f16 transposed ----
  k_convert_transpose<<<cdiv(IN * HID, T), T, 0, stream>>>(W_in, WTin, IN, HID);
  for (int l = 0; l < 3; ++l) {
    k_convert_transpose<<<cdiv(HID * HID, T), T, 0, stream>>>(W_src + (size_t)l * HID * HID,
                                                              WsT   + (size_t)l * HID * HID, HID, HID);
    k_convert_transpose<<<cdiv(HID * HID, T), T, 0, stream>>>(W_dst + (size_t)l * HID * HID,
                                                              WdT   + (size_t)l * HID * HID, HID, HID);
  }

  // ---- input projection: h = x @ W_in + b_in (also emits h16 for layer 0) ----
  k_f32_to_f16<<<cdiv(N * IN, T), T, 0, stream>>>(node_feats, x16, N * IN);
  k_gemm_wmma<64, true><<<cdiv(N, 16), 256, 0, stream>>>(x16, WTin, b_in, h, h16, N);

  // ---- 3 GATv2 layers ----
  const int edgeBlocks = cdiv(EP, 8);
  for (int l = 0; l < 3; ++l) {
    k_gemm_wmma<128, false><<<cdiv(N, 16), 256, 0, stream>>>(h16, WsT + (size_t)l * HID * HID,
                                                             b_src + (size_t)l * HID, fs, nullptr, N);
    k_gemm_wmma<128, false><<<cdiv(N, 16), 256, 0, stream>>>(h16, WdT + (size_t)l * HID * HID,
                                                             b_dst + (size_t)l * HID, fd, nullptr, N);
    k_init_mz4<<<cdiv(N * HEADS / 4, T), T, 0, stream>>>((float4*)mbuf, (float4*)zbuf, N * HEADS / 4);
    k_copy_f32x4<<<cdiv(N * HID / 4, T), T, 0, stream>>>((const float4*)h, (float4*)agg,
                                                         N * HID / 4);   // residual inside gatv2
    k_edge_logits<<<edgeBlocks, 256, 0, stream>>>(fs, fd, attn + (size_t)l * HID,
                                                  src, dst, logits, mbuf, E, EP);
    k_edge_exp<<<cdiv(EP, T), T, 0, stream>>>((float4*)logits, (const float4*)mbuf,
                                              zbuf, dst, E, EP);
    k_edge_scatter<<<edgeBlocks, 256, 0, stream>>>(fs, logits, zbuf, src, dst, agg, E, EP);
    k_ln_relu<<<cdiv(N, 8), 256, 0, stream>>>(h, agg, ln_g + (size_t)l * HID,
                                              ln_b + (size_t)l * HID, h16, N);
  }

  // ---- outputs: [graph_emb(128) | h(N*128)] ----
  k_copy_f32x4<<<cdiv(N * HID / 4, T), T, 0, stream>>>((const float4*)h, (float4*)(out + HID),
                                                       N * HID / 4);
  k_colmean<<<HID, 256, 0, stream>>>(h, out, N);
}
